// Attention_66477503808179
// MI455X (gfx1250) — matrix operand order repaired and checked by reading, not compiled, not run on hardware
//
#include <hip/hip_runtime.h>

// ---------------------------------------------------------------------------
// Problem constants (from reference)
// ---------------------------------------------------------------------------
#define DIMN 4096
#define NH   32
#define HD   128
#define BSZ  2
#define SQ   1024
#define EL   512
#define KVL  1536   // EL + SQ

// Use gfx1250 async global->LDS DMA (ASYNCcnt) instead of load+ds_store.
#define USE_ASYNC 1

typedef __bf16 bf16_t;
typedef bf16_t v16bf __attribute__((ext_vector_type(16)));
typedef float  v8f   __attribute__((ext_vector_type(8)));
typedef unsigned int u32x4 __attribute__((ext_vector_type(4)));

union FragU { u32x4 u[2]; v16bf v; };

__device__ __forceinline__ unsigned short f2bfu(float f) {
    unsigned u = __builtin_bit_cast(unsigned, f);
    u += 0x7FFFu + ((u >> 16) & 1u);          // round-to-nearest-even
    return (unsigned short)(u >> 16);
}

// Async copy of 16 bytes from global to LDS. The LDS-address VGPR holds the
// wave-relative byte offset (low 32 bits of the flat LDS address).
__device__ __forceinline__ void async_ld_b128(const unsigned short* gptr,
                                              unsigned short* lptr) {
#if USE_ASYNC
    unsigned lds_off = (unsigned)(uintptr_t)lptr;
    asm volatile("global_load_async_to_lds_b128 %0, %1, off"
                 :: "v"(lds_off), "v"(gptr) : "memory");
#else
    *(u32x4*)lptr = *(const u32x4*)gptr;
#endif
}

__device__ __forceinline__ void async_wait_all() {
#if USE_ASYNC
    asm volatile("s_wait_asynccnt 0x0" ::: "memory");
#endif
}

// ---------------------------------------------------------------------------
// fp32 -> bf16 conversion (grid-stride)
// ---------------------------------------------------------------------------
__global__ void k_cvt(const float* __restrict__ in, unsigned short* __restrict__ out, int n) {
    for (int i = blockIdx.x * blockDim.x + threadIdx.x; i < n; i += gridDim.x * blockDim.x)
        out[i] = f2bfu(in[i]);
}

// ---------------------------------------------------------------------------
// C[M x N] (fp32) = A[M x K] (bf16, row-major) @ B[N x K]^T (bf16, row-major)
// Block: 256 threads = 8 waves; tile 128x128; BK=32. Wave grid 4(M) x 2(N),
// each wave computes 32x64 = 2x4 WMMA 16x16 accumulators.
// Double-buffered LDS, staged by async global->LDS DMA.
// ---------------------------------------------------------------------------
__global__ __launch_bounds__(256)
void k_gemm_bf16(const unsigned short* __restrict__ A,
                 const unsigned short* __restrict__ B,
                 float* __restrict__ C, int M, int N, int K)
{
    __shared__ unsigned short As[2][128 * 32];
    __shared__ unsigned short Bs[2][128 * 32];

    const int tid  = threadIdx.x;
    const int wv   = tid >> 5;
    const int L    = tid & 31;
    const int hh   = L >> 4;     // lane half
    const int n16  = L & 15;
    const int wm   = wv >> 1;    // 0..3
    const int wn   = wv & 1;     // 0..1
    const int rowBase = blockIdx.y * 128;
    const int colBase = blockIdx.x * 128;

    auto stage = [&](int kt, int buf) {
#pragma unroll
        for (int i = 0; i < 2; ++i) {
            int c  = tid + 256 * i;        // 512 chunks of 8 bf16
            int r  = c >> 2;               // 0..127
            int c8 = (c & 3) * 8;          // 0,8,16,24
            async_ld_b128(A + (size_t)(rowBase + r) * K + kt * 32 + c8,
                          &As[buf][r * 32 + c8]);
            async_ld_b128(B + (size_t)(colBase + r) * K + kt * 32 + c8,
                          &Bs[buf][r * 32 + c8]);
        }
    };

    v8f zz = {};
    v8f acc[2][4];
#pragma unroll
    for (int i = 0; i < 2; ++i)
#pragma unroll
        for (int j = 0; j < 4; ++j) acc[i][j] = zz;

    const int kTiles = K >> 5;
    stage(0, 0);
    for (int kt = 0; kt < kTiles; ++kt) {
        const int cur = kt & 1;
        async_wait_all();          // this wave's tile-kt DMA complete
        __syncthreads();           // -> everyone's tile-kt DMA complete
        if (kt + 1 < kTiles) stage(kt + 1, cur ^ 1);

        v16bf afr[2], bfr[4];
#pragma unroll
        for (int mi = 0; mi < 2; ++mi) {
            const unsigned short* p = &As[cur][(wm * 32 + mi * 16 + n16) * 32 + hh * 8];
            FragU t;
            t.u[0] = *(const u32x4*)(p);        // K = hh*8 + 0..7
            t.u[1] = *(const u32x4*)(p + 16);   // K = 16 + hh*8 + 0..7
            afr[mi] = t.v;
        }
#pragma unroll
        for (int ni = 0; ni < 4; ++ni) {
            const unsigned short* p = &Bs[cur][(wn * 64 + ni * 16 + n16) * 32 + hh * 8];
            FragU t;
            t.u[0] = *(const u32x4*)(p);        // K = hh*8 + 0..7
            t.u[1] = *(const u32x4*)(p + 16);   // K = 16 + hh*8 + 0..7
            bfr[ni] = t.v;
        }
#pragma unroll
        for (int mi = 0; mi < 2; ++mi)
#pragma unroll
            for (int ni = 0; ni < 4; ++ni)
                acc[mi][ni] = __builtin_amdgcn_wmma_f32_16x16x32_bf16(
                    false, afr[mi], false, bfr[ni], (short)0, acc[mi][ni], false, false);
    }

#pragma unroll
    for (int mi = 0; mi < 2; ++mi)
#pragma unroll
        for (int ni = 0; ni < 4; ++ni) {
            int gn = colBase + wn * 64 + ni * 16 + n16;
#pragma unroll
            for (int r = 0; r < 8; ++r) {
                int gm = rowBase + wm * 32 + mi * 16 + hh * 8 + r;
                C[(size_t)gm * N + gn] = acc[mi][ni][r];
            }
        }
}

// ---------------------------------------------------------------------------
// RoPE on Q (with 1/sqrt(HD) folded in) and self-K; outputs bf16.
// One thread per (row, pair). Kc row offset = b*KVL + EL + s.
// ---------------------------------------------------------------------------
__global__ void k_rope(const float* __restrict__ Q32, const float* __restrict__ Kx32,
                       const float* __restrict__ cosb, const float* __restrict__ sinb,
                       unsigned short* __restrict__ Qb, unsigned short* __restrict__ Kc)
{
    int idx = blockIdx.x * blockDim.x + threadIdx.x;
    const int total = BSZ * SQ * (DIMN / 2);
    if (idx >= total) return;
    int p   = idx % (DIMN / 2);
    int row = idx / (DIMN / 2);       // b*SQ + s
    int s   = row % SQ, b = row / SQ;
    int head = p / (HD / 2);
    int i    = p % (HD / 2);
    int col  = head * HD + 2 * i;
    float c  = cosb[s * (HD / 2) + i];
    float sn = sinb[s * (HD / 2) + i];

    const float qs = 0.08838834764831845f;     // 1/sqrt(128)
    float qa = Q32[(size_t)row * DIMN + col];
    float qb = Q32[(size_t)row * DIMN + col + 1];
    Qb[(size_t)row * DIMN + col]     = f2bfu((qa * c - qb * sn) * qs);
    Qb[(size_t)row * DIMN + col + 1] = f2bfu((qa * sn + qb * c) * qs);

    float ka = Kx32[(size_t)row * DIMN + col];
    float kb = Kx32[(size_t)row * DIMN + col + 1];
    size_t krow = (size_t)b * KVL + EL + s;
    Kc[krow * DIMN + col]     = f2bfu(ka * c - kb * sn);
    Kc[krow * DIMN + col + 1] = f2bfu(ka * sn + kb * c);
}

// ---------------------------------------------------------------------------
// Build enc-K (no RoPE) and full V cache (bf16), layout [b, KVL, DIMN].
// ---------------------------------------------------------------------------
__global__ void k_build_cache(const float* __restrict__ Ke32, const float* __restrict__ Ve32,
                              const float* __restrict__ Vx32,
                              unsigned short* __restrict__ Kc, unsigned short* __restrict__ Vc)
{
    int idx = blockIdx.x * blockDim.x + threadIdx.x;
    const int total = BSZ * KVL * DIMN;
    if (idx >= total) return;
    int col = idx % DIMN;
    int row = idx / DIMN;             // b*KVL + s
    int s   = row % KVL, b = row / KVL;
    if (s < EL) {
        size_t er = (size_t)(b * EL + s) * DIMN + col;
        Kc[(size_t)row * DIMN + col] = f2bfu(Ke32[er]);
        Vc[(size_t)row * DIMN + col] = f2bfu(Ve32[er]);
    } else {
        size_t xr = (size_t)(b * SQ + (s - EL)) * DIMN + col;
        Vc[(size_t)row * DIMN + col] = f2bfu(Vx32[xr]);
    }
}

// ---------------------------------------------------------------------------
// Flash attention. Grid: (SQ/128, NH, BSZ); 256 threads = 8 waves,
// each wave owns 16 q-rows. K tile = 64 keys. All matmuls via WMMA bf16.
// K tile staged by async DMA; V tile transposed manually into LDS.
// ---------------------------------------------------------------------------
__global__ __launch_bounds__(256)
void k_attn(const unsigned short* __restrict__ Qb,
            const unsigned short* __restrict__ Kc,
            const unsigned short* __restrict__ Vc,
            unsigned short* __restrict__ Ob)
{
    __shared__ unsigned short Ks[64 * 128];        // K tile   [key][d]
    __shared__ unsigned short Vt[128 * 64];        // V tile^T [d][key]
    __shared__ unsigned short Ps[8 * 16 * 64];     // per-wave P scratch

    const int tid = threadIdx.x;
    const int wv  = tid >> 5, L = tid & 31, hh = L >> 4, n16 = L & 15;
    const int qi   = blockIdx.x;
    const int head = blockIdx.y;
    const int b    = blockIdx.z;
    const int qr   = qi * 128 + wv * 16;

    // Q fragments (A-layout), 4 k-steps over d=128
    v16bf aQ[4];
    const unsigned short* qbase = Qb + ((size_t)(b * SQ + qr + n16)) * DIMN + head * HD;
#pragma unroll
    for (int ks = 0; ks < 4; ++ks) {
        FragU t;
        t.u[0] = *(const u32x4*)(qbase + ks * 32 + hh * 8);
        t.u[1] = *(const u32x4*)(qbase + ks * 32 + 16 + hh * 8);
        aQ[ks] = t.v;
    }

    v8f zz = {};
    v8f o[8];
#pragma unroll
    for (int j = 0; j < 8; ++j) o[j] = zz;
    float rmax[8], rsum[8];
#pragma unroll
    for (int r = 0; r < 8; ++r) { rmax[r] = -1e30f; rsum[r] = 0.0f; }

    const unsigned short* kbase = Kc + (size_t)b * KVL * DIMN + head * HD;
    const unsigned short* vbase = Vc + (size_t)b * KVL * DIMN + head * HD;
    const int ktEnd = 10 + 2 * qi;   // tiles beyond causal horizon fully masked

    for (int kt = 0; kt < ktEnd; ++kt) {
        __syncthreads();
        // ---- stage K (async DMA) and V (manual transpose) tiles ----
#pragma unroll
        for (int i = 0; i < 4; ++i) {
            int c  = tid + 256 * i;      // 1024 chunks of 8
            int r  = c >> 4;             // key_local 0..63
            int c8 = (c & 15) * 8;       // d 0..120
            async_ld_b128(kbase + (size_t)(kt * 64 + r) * DIMN + c8, Ks + r * 128 + c8);
            u32x4 vv = *(const u32x4*)(vbase + (size_t)(kt * 64 + r) * DIMN + c8);
            unsigned short tmp[8];
            *(u32x4*)tmp = vv;
#pragma unroll
            for (int e = 0; e < 8; ++e)
                Vt[(c8 + e) * 64 + r] = tmp[e];
        }
        async_wait_all();
        __syncthreads();

        // ---- S = Q @ K^T (16 WMMA) ----
        v8f s[4];
#pragma unroll
        for (int j = 0; j < 4; ++j) s[j] = zz;
#pragma unroll
        for (int ks = 0; ks < 4; ++ks) {
#pragma unroll
            for (int j = 0; j < 4; ++j) {
                const unsigned short* p = Ks + (j * 16 + n16) * 128 + ks * 32 + hh * 8;
                FragU t;
                t.u[0] = *(const u32x4*)(p);
                t.u[1] = *(const u32x4*)(p + 16);
                s[j] = __builtin_amdgcn_wmma_f32_16x16x32_bf16(
                    false, aQ[ks], false, t.v, (short)0, s[j], false, false);
            }
        }

        // ---- mask + online softmax ----
#pragma unroll
        for (int r = 0; r < 8; ++r) {
            int q = qi * 128 + wv * 16 + hh * 8 + r;
            float mx = -1e30f;
#pragma unroll
            for (int j = 0; j < 4; ++j) {
                int key = kt * 64 + j * 16 + n16;
                float sv = s[j][r];
                if (key >= EL && (key - EL) > q) sv = -1e9f;
                s[j][r] = sv;
                mx = fmaxf(mx, sv);
            }
            mx = fmaxf(mx, __shfl_xor(mx, 1, 32));
            mx = fmaxf(mx, __shfl_xor(mx, 2, 32));
            mx = fmaxf(mx, __shfl_xor(mx, 4, 32));
            mx = fmaxf(mx, __shfl_xor(mx, 8, 32));
            float newm  = fmaxf(rmax[r], mx);
            float alpha = __expf(rmax[r] - newm);
            rmax[r] = newm;
            float ps = 0.0f;
#pragma unroll
            for (int j = 0; j < 4; ++j) {
                float pj = __expf(s[j][r] - newm);
                s[j][r] = pj;
                ps += pj;
            }
            ps += __shfl_xor(ps, 1, 32);
            ps += __shfl_xor(ps, 2, 32);
            ps += __shfl_xor(ps, 4, 32);
            ps += __shfl_xor(ps, 8, 32);
            rsum[r] = rsum[r] * alpha + ps;
#pragma unroll
            for (int jd = 0; jd < 8; ++jd) o[jd][r] *= alpha;
        }

        // ---- C-layout -> A-layout conversion of P via per-wave LDS ----
        unsigned short* pw = Ps + wv * (16 * 64);
#pragma unroll
        for (int r = 0; r < 8; ++r)
#pragma unroll
            for (int j = 0; j < 4; ++j)
                pw[(hh * 8 + r) * 64 + j * 16 + n16] = f2bfu(s[j][r]);
        // wave-local LDS region: same-wave DS ops are in-order; no barrier needed

        // ---- O += P @ V (16 WMMA) ----
#pragma unroll
        for (int ks2 = 0; ks2 < 2; ++ks2) {
            FragU ta;
            const unsigned short* pa = pw + n16 * 64 + ks2 * 32 + hh * 8;
            ta.u[0] = *(const u32x4*)(pa);
            ta.u[1] = *(const u32x4*)(pa + 16);
#pragma unroll
            for (int jd = 0; jd < 8; ++jd) {
                const unsigned short* pv = Vt + (jd * 16 + n16) * 64 + ks2 * 32 + hh * 8;
                FragU tb;
                tb.u[0] = *(const u32x4*)(pv);
                tb.u[1] = *(const u32x4*)(pv + 16);
                o[jd] = __builtin_amdgcn_wmma_f32_16x16x32_bf16(
                    false, ta.v, false, tb.v, (short)0, o[jd], false, false);
            }
        }
    }

    // ---- normalize + store bf16 attn output [b*SQ + q][head*HD + d] ----
#pragma unroll
    for (int jd = 0; jd < 8; ++jd)
#pragma unroll
        for (int r = 0; r < 8; ++r) {
            int q = qi * 128 + wv * 16 + hh * 8 + r;
            float val = o[jd][r] / rsum[r];
            Ob[((size_t)(b * SQ + q)) * DIMN + head * HD + jd * 16 + n16] = f2bfu(val);
        }
}

// ---------------------------------------------------------------------------
// Host launcher
// ---------------------------------------------------------------------------
extern "C" void kernel_launch(void* const* d_in, const int* in_sizes, int n_in,
                              void* d_out, int out_size, void* d_ws, size_t ws_size,
                              hipStream_t stream)
{
    const float* x    = (const float*)d_in[0];
    const float* fcos = (const float*)d_in[2];
    const float* fsin = (const float*)d_in[3];
    const float* enc  = (const float*)d_in[5];
    const float* wq   = (const float*)d_in[6];
    const float* wk   = (const float*)d_in[7];
    const float* wv   = (const float*)d_in[8];
    const float* wo   = (const float*)d_in[9];
    float* out = (float*)d_out;

    char* ws = (char*)d_ws;
    size_t off = 0;
    auto alloc = [&](size_t bytes) -> char* {
        char* p = ws + off;
        off = (off + bytes + 255) & ~(size_t)255;
        return p;
    };
    const size_t WE = (size_t)DIMN * DIMN;                 // 16M
    const size_t XR = (size_t)BSZ * SQ;                    // 2048 rows
    const size_t ER = (size_t)BSZ * EL;                    // 1024 rows
    const size_t KR = (size_t)BSZ * KVL;                   // 3072 rows

    unsigned short* Wqb = (unsigned short*)alloc(WE * 2);
    unsigned short* Wkb = (unsigned short*)alloc(WE * 2);
    unsigned short* Wvb = (unsigned short*)alloc(WE * 2);
    unsigned short* Wob = (unsigned short*)alloc(WE * 2);
    unsigned short* Xb  = (unsigned short*)alloc(XR * DIMN * 2);
    unsigned short* Eb  = (unsigned short*)alloc(ER * DIMN * 2);
    float* Q32  = (float*)alloc(XR * DIMN * 4);
    float* Kx32 = (float*)alloc(XR * DIMN * 4);
    float* Vx32 = (float*)alloc(XR * DIMN * 4);
    float* Ke32 = (float*)alloc(ER * DIMN * 4);
    float* Ve32 = (float*)alloc(ER * DIMN * 4);
    unsigned short* Qbf = (unsigned short*)alloc(XR * DIMN * 2);
    unsigned short* Kc  = (unsigned short*)alloc(KR * DIMN * 2);
    unsigned short* Vc  = (unsigned short*)alloc(KR * DIMN * 2);
    unsigned short* AOb = (unsigned short*)alloc(XR * DIMN * 2);

    const int T = 256;
    // 1) convert weights + activations to bf16
    k_cvt<<<4096, T, 0, stream>>>(wq,  Wqb, (int)WE);
    k_cvt<<<4096, T, 0, stream>>>(wk,  Wkb, (int)WE);
    k_cvt<<<4096, T, 0, stream>>>(wv,  Wvb, (int)WE);
    k_cvt<<<4096, T, 0, stream>>>(wo,  Wob, (int)WE);
    k_cvt<<<4096, T, 0, stream>>>(x,   Xb,  (int)(XR * DIMN));
    k_cvt<<<4096, T, 0, stream>>>(enc, Eb,  (int)(ER * DIMN));

    // 2) projection GEMMs (C = A @ W^T)
    dim3 gX(DIMN / 128, (unsigned)(XR / 128));   // 32 x 16
    dim3 gE(DIMN / 128, (unsigned)(ER / 128));   // 32 x 8
    k_gemm_bf16<<<gX, T, 0, stream>>>(Xb, Wqb, Q32,  (int)XR, DIMN, DIMN);
    k_gemm_bf16<<<gX, T, 0, stream>>>(Xb, Wkb, Kx32, (int)XR, DIMN, DIMN);
    k_gemm_bf16<<<gX, T, 0, stream>>>(Xb, Wvb, Vx32, (int)XR, DIMN, DIMN);
    k_gemm_bf16<<<gE, T, 0, stream>>>(Eb, Wkb, Ke32, (int)ER, DIMN, DIMN);
    k_gemm_bf16<<<gE, T, 0, stream>>>(Eb, Wvb, Ve32, (int)ER, DIMN, DIMN);

    // 3) RoPE + cache build
    int ropeTot = BSZ * SQ * (DIMN / 2);
    k_rope<<<(ropeTot + T - 1) / T, T, 0, stream>>>(Q32, Kx32, fcos, fsin, Qbf, Kc);
    int cacheTot = BSZ * KVL * DIMN;
    k_build_cache<<<(cacheTot + T - 1) / T, T, 0, stream>>>(Ke32, Ve32, Vx32, Kc, Vc);

    // 4) flash attention
    dim3 gA(SQ / 128, NH, BSZ);
    k_attn<<<gA, T, 0, stream>>>(Qbf, Kc, Vc, AOb);

    // 5) output projection straight into d_out (fp32)
    k_gemm_bf16<<<gX, T, 0, stream>>>(AOb, Wob, out, (int)XR, DIMN, DIMN);
}
